// BlocksCore_25683904430710
// MI455X (gfx1250) — matrix operand order repaired and checked by reading, not compiled, not run on hardware
//
#include <hip/hip_runtime.h>
#include <hip/hip_bf16.h>

#define B_    4096
#define NINP  512
#define NHID  2048
#define NBO   8
#define TOPK  4
#define BSO   256
#define DK_I  64

typedef unsigned short u16;
typedef unsigned int   u32;
typedef __attribute__((ext_vector_type(16))) __bf16 v16bf;
typedef __attribute__((ext_vector_type(8)))  float  v8f;

union FragB   { u32 u[8];  v16bf v; };
union FragAcc { float f[8]; v8f  v; };

__device__ __forceinline__ u16 f2bf(float f) {
  u32 u = __float_as_uint(f);
  u32 r = u + 0x7FFFu + ((u >> 16) & 1u);
  return (u16)(r >> 16);
}

__device__ __forceinline__ v8f zerov8() {
  v8f v;
#pragma unroll
  for (int i = 0; i < 8; ++i) v[i] = 0.f;
  return v;
}

// ---------------- conversion kernels ----------------
__global__ void k_cvt(const float* __restrict__ in, u16* __restrict__ out, int n) {
  int i = blockIdx.x * blockDim.x + threadIdx.x;
  if (i < n) out[i] = f2bf(in[i]);
}

// in: K x N row-major  ->  out: N x K row-major (bf16)
__global__ void k_cvt_t(const float* __restrict__ in, u16* __restrict__ out, int K, int N) {
  int i = blockIdx.x * blockDim.x + threadIdx.x;
  if (i < K * N) {
    int n = i % N, k = i / N;
    out[(long)n * K + k] = f2bf(in[i]);
  }
}

// Wq_m/Wk_m/Wv_m (8,256,64) -> wm[k][j*256+i], j:0..63 q, 64..127 k, 128..191 v
__global__ void k_build_wm(const float* __restrict__ q, const float* __restrict__ kk,
                           const float* __restrict__ v, u16* __restrict__ out) {
  int idx = blockIdx.x * blockDim.x + threadIdx.x;
  if (idx >= NBO * 192 * 256) return;
  int i = idx % 256;
  int j = (idx / 256) % 192;
  int blk = idx / (256 * 192);
  const float* src = (j < 64) ? q : (j < 128 ? kk : v);
  int o = j & 63;
  out[idx] = f2bf(src[(long)blk * 256 * 64 + (long)i * 64 + o]);
}

// c[k,o] = bih[k,o] + dot(fc_i_b, Wih[k,o,:])
__global__ void k_cko(const float* __restrict__ Wih, const float* __restrict__ bih,
                      const float* __restrict__ fcib, float* __restrict__ c) {
  int idx = blockIdx.x * blockDim.x + threadIdx.x;
  if (idx >= NBO * 768) return;
  const float* w = Wih + (long)idx * 1024;
  float acc = bih[idx];
  for (int i = 0; i < 1024; ++i) acc += fcib[i] * w[i];
  c[idx] = acc;
}

// ---------------- k1 = inp @ Wk_i[1]  (B x 64) ----------------
__global__ void k_k1(const float* __restrict__ inp, const float* __restrict__ Wk,
                     float* __restrict__ k1) {
  int idx = blockIdx.x * blockDim.x + threadIdx.x;
  if (idx >= B_ * 64) return;
  int d = idx & 63;
  int b = idx >> 6;
  const float* x = inp + (long)b * NINP;
  const float* w = Wk + (long)NINP * DK_I + d;   // second matrix of Wk_i
  float acc = 0.f;
  for (int i = 0; i < NINP; ++i) acc += x[i] * w[(long)i * DK_I];
  k1[idx] = acc;
}

// ---------------- scores, iatt1, mask ----------------
__global__ void k_score(const float* __restrict__ hx, const float* __restrict__ Wq,
                        const float* __restrict__ k1, float* __restrict__ iatt1,
                        float* __restrict__ masko) {
  __shared__ float s_sh[256];
  __shared__ float k1_sh[32 * 64];
  int tid = threadIdx.x;
  int bq = blockIdx.x * 32;
  for (int i = tid; i < 32 * 64; i += 256)
    k1_sh[i] = k1[(long)(bq + (i >> 6)) * 64 + (i & 63)];
  __syncthreads();
  int b = bq + (tid >> 3);
  int n = tid & 7;
  const float* h = hx + (long)b * NHID + n * BSO;
  const float* w = Wq + (long)n * BSO * DK_I;
  const float* kv = k1_sh + (tid >> 3) * 64;
  float s = 0.f;
  for (int d = 0; d < DK_I; ++d) {
    float qd = 0.f;
    for (int i = 0; i < BSO; ++i) qd += h[i] * w[(long)i * DK_I + d];
    s += qd * kv[d];
  }
  s *= 0.125f;  // 1/sqrt(64)
  s_sh[tid] = s;
  __syncthreads();
  const float* grp = s_sh + (tid >> 3) * 8;
  int cnt = 0;
#pragma unroll
  for (int m = 0; m < 8; ++m)
    if (grp[m] < s || (grp[m] == s && m < n)) cnt++;
  masko[(long)b * 8 + n] = (cnt >= (NBO - TOPK)) ? 1.f : 0.f;
  iatt1[(long)b * 8 + n] = 1.f / (1.f + __expf(-s));
}

// ---------------- generic bf16 WMMA GEMM: C[m,n] = sum_k A[m,k]*Bt[n,k] ----------------
// Register-blocked: each wave computes a (16*TM) x (16*TN) C tile.
#define TM 4
#define TN 2
__global__ void k_gemm(const u16* __restrict__ A, int lda, int abo,
                       const u16* __restrict__ Bt, long bstride, int ldb,
                       float* __restrict__ Cf, u16* __restrict__ Cb,
                       int ldc, int cbo, int ksteps) {
  int lane = threadIdx.x & 31;
  int wave = threadIdx.x >> 5;
  int z = blockIdx.z;
  int half = lane >> 4;
  int lid = lane & 15;
  int row_base = blockIdx.x * (16 * TM);
  int col_base = (blockIdx.y * (blockDim.x >> 5) + wave) * (16 * TN);

  const u16* Ab = A + (long)z * abo;
  const u16* Bb = Bt + (long)z * bstride;

  v8f acc[TM][TN];
#pragma unroll
  for (int r = 0; r < TM; ++r)
#pragma unroll
    for (int c = 0; c < TN; ++c) acc[r][c] = zerov8();

  for (int ks = 0; ks < ksteps; ++ks) {
    int kbase = ks * 32;
    // L2 prefetch a few k-steps ahead (global_prefetch_b8)
    if (ks + 4 < ksteps) {
      __builtin_prefetch(Ab + (long)(row_base + lid) * lda + (ks + 4) * 32, 0, 1);
      __builtin_prefetch(Bb + (long)(col_base + lid) * ldb + (ks + 4) * 32, 0, 1);
    }
    FragB bfr[TN];
#pragma unroll
    for (int c = 0; c < TN; ++c) {
      const u32* p = (const u32*)(Bb + (long)(col_base + c * 16 + lid) * ldb + kbase + (half ? 16 : 0));
#pragma unroll
      for (int v = 0; v < 8; ++v) bfr[c].u[v] = p[v];
    }
    FragB afr[TM];
#pragma unroll
    for (int r = 0; r < TM; ++r) {
      long abase = (long)(row_base + r * 16 + lid) * lda + kbase + (half ? 16 : 0);
      const u32* p0 = (const u32*)(Ab + abase);
      const u32* p1 = (const u32*)(Ab + abase + 8);
      afr[r].u[0] = p0[0]; afr[r].u[1] = p0[1]; afr[r].u[2] = p0[2]; afr[r].u[3] = p0[3];
      afr[r].u[4] = p1[0]; afr[r].u[5] = p1[1]; afr[r].u[6] = p1[2]; afr[r].u[7] = p1[3];
    }
#pragma unroll
    for (int r = 0; r < TM; ++r)
#pragma unroll
      for (int c = 0; c < TN; ++c)
        acc[r][c] = __builtin_amdgcn_wmma_f32_16x16x32_bf16(
            false, afr[r].v, false, bfr[c].v, (short)0, acc[r][c], false, false);
  }

#pragma unroll
  for (int r = 0; r < TM; ++r)
#pragma unroll
    for (int c = 0; c < TN; ++c) {
      FragAcc fa; fa.v = acc[r][c];
#pragma unroll
      for (int v = 0; v < 8; ++v) {
        int rowm = row_base + r * 16 + v + half * 8;
        int col = col_base + c * 16 + lid;
        long idx = (long)rowm * ldc + (long)z * cbo + col;
        if (Cf) Cf[idx] = fa.f[v];
        if (Cb) Cb[idx] = f2bf(fa.f[v]);
      }
    }
}

// ---------------- fused GRU: gx (K=1024) + gh (K=256) + gates ----------------
// grid: (B/16, NBO), block 256 (8 waves). Wave w handles cols {g*256 + w*32 .. +32} for g=r,z,n.
__global__ void k_gru(const u16* __restrict__ U, const u16* __restrict__ Hb,
                      const u16* __restrict__ Wih, const u16* __restrict__ Whh,
                      const float* __restrict__ cko, const float* __restrict__ bhh,
                      const float* __restrict__ iatt1, const float* __restrict__ hx,
                      float* __restrict__ hnew, u16* __restrict__ hnewb) {
  int lane = threadIdx.x & 31;
  int wave = threadIdx.x >> 5;
  int half = lane >> 4;
  int lid = lane & 15;
  int row_base = blockIdx.x * 16;
  int k = blockIdx.y;

  v8f accx[6], acch[6];
#pragma unroll
  for (int t = 0; t < 6; ++t) { accx[t] = zerov8(); acch[t] = zerov8(); }

  const u16* Wk = Wih + (long)k * 768 * 1024;
  for (int ks = 0; ks < 32; ++ks) {       // gx: A = u, K = 1024
    int kbase = ks * 32;
    if (ks + 4 < 32)
      __builtin_prefetch(U + (long)(row_base + lid) * 1024 + (ks + 4) * 32, 0, 1);
    FragB afr;
    {
      long abase = (long)(row_base + lid) * 1024 + kbase + (half ? 16 : 0);
      const u32* p0 = (const u32*)(U + abase);
      const u32* p1 = (const u32*)(U + abase + 8);
      afr.u[0] = p0[0]; afr.u[1] = p0[1]; afr.u[2] = p0[2]; afr.u[3] = p0[3];
      afr.u[4] = p1[0]; afr.u[5] = p1[1]; afr.u[6] = p1[2]; afr.u[7] = p1[3];
    }
#pragma unroll
    for (int t = 0; t < 6; ++t) {
      int col = (t >> 1) * 256 + wave * 32 + (t & 1) * 16 + lid;
      const u32* bp = (const u32*)(Wk + (long)col * 1024 + kbase + (half ? 16 : 0));
      FragB bfr;
#pragma unroll
      for (int v = 0; v < 8; ++v) bfr.u[v] = bp[v];
      accx[t] = __builtin_amdgcn_wmma_f32_16x16x32_bf16(
          false, afr.v, false, bfr.v, (short)0, accx[t], false, false);
    }
  }

  const u16* Wh = Whh + (long)k * 768 * 256;
  for (int ks = 0; ks < 8; ++ks) {        // gh: A = hx block k, K = 256
    int kbase = ks * 32;
    FragB afr;
    {
      long abase = (long)(row_base + lid) * 2048 + k * 256 + kbase + (half ? 16 : 0);
      const u32* p0 = (const u32*)(Hb + abase);
      const u32* p1 = (const u32*)(Hb + abase + 8);
      afr.u[0] = p0[0]; afr.u[1] = p0[1]; afr.u[2] = p0[2]; afr.u[3] = p0[3];
      afr.u[4] = p1[0]; afr.u[5] = p1[1]; afr.u[6] = p1[2]; afr.u[7] = p1[3];
    }
#pragma unroll
    for (int t = 0; t < 6; ++t) {
      int col = (t >> 1) * 256 + wave * 32 + (t & 1) * 16 + lid;
      const u32* bp = (const u32*)(Wh + (long)col * 256 + kbase + (half ? 16 : 0));
      FragB bfr;
#pragma unroll
      for (int v = 0; v < 8; ++v) bfr.u[v] = bp[v];
      acch[t] = __builtin_amdgcn_wmma_f32_16x16x32_bf16(
          false, afr.v, false, bfr.v, (short)0, acch[t], false, false);
    }
  }

#pragma unroll
  for (int tt = 0; tt < 2; ++tt) {
    int colr = wave * 32 + tt * 16 + lid;          // 0..255 within block
    float cr = cko[k * 768 + colr];
    float cz = cko[k * 768 + 256 + colr];
    float cn = cko[k * 768 + 512 + colr];
    float br = bhh[k * 768 + colr];
    float bz = bhh[k * 768 + 256 + colr];
    float bn = bhh[k * 768 + 512 + colr];
    FragAcc xr_, xz_, xn_, hr_, hz_, hn_;
    xr_.v = accx[tt];     xz_.v = accx[2 + tt]; xn_.v = accx[4 + tt];
    hr_.v = acch[tt];     hz_.v = acch[2 + tt]; hn_.v = acch[4 + tt];
#pragma unroll
    for (int v = 0; v < 8; ++v) {
      int row = row_base + v + half * 8;
      float ia  = iatt1[(long)row * 8 + k];
      float gxr = ia * xr_.f[v] + cr;
      float gxz = ia * xz_.f[v] + cz;
      float gxn = ia * xn_.f[v] + cn;
      float ghr = hr_.f[v] + br;
      float ghz = hz_.f[v] + bz;
      float ghn = hn_.f[v] + bn;
      float r  = 1.f / (1.f + __expf(-(gxr + ghr)));
      float zg = 1.f / (1.f + __expf(-(gxz + ghz)));
      float nn = tanhf(gxn + r * ghn);
      long idx = (long)row * NHID + k * 256 + colr;
      float hv = (1.f - zg) * nn + zg * hx[idx];
      hnew[idx]  = hv;
      hnewb[idx] = f2bf(hv);
    }
  }
}

// ---------------- mini attention over 8 blocks, 4 heads x d16: one lane per (h,q) ----------------
__global__ void k_mattn(const float* __restrict__ qkv, float* __restrict__ om) {
  int lane = threadIdx.x & 31;
  int b = blockIdx.x * (blockDim.x >> 5) + (threadIdx.x >> 5);
  int h = lane >> 3, q = lane & 7;
  const float* base = qkv + (long)b * (NBO * 192);
  float qv[16];
#pragma unroll
  for (int d = 0; d < 16; ++d) qv[d] = base[q * 192 + h * 16 + d];
  float lg[8];
  float mx = -1e30f;
#pragma unroll
  for (int kb = 0; kb < 8; ++kb) {
    float s = 0.f;
#pragma unroll
    for (int d = 0; d < 16; ++d) s += qv[d] * base[kb * 192 + 64 + h * 16 + d];
    s *= 0.25f;  // 1/sqrt(16)
    lg[kb] = s;
    mx = fmaxf(mx, s);
  }
  float den = 0.f;
#pragma unroll
  for (int kb = 0; kb < 8; ++kb) { lg[kb] = __expf(lg[kb] - mx); den += lg[kb]; }
  float inv = 1.f / den;
  float o[16];
#pragma unroll
  for (int d = 0; d < 16; ++d) o[d] = 0.f;
#pragma unroll
  for (int kb = 0; kb < 8; ++kb) {
    float a = lg[kb] * inv;
#pragma unroll
    for (int d = 0; d < 16; ++d) o[d] += a * base[kb * 192 + 128 + h * 16 + d];
  }
#pragma unroll
  for (int d = 0; d < 16; ++d) om[(long)b * 512 + q * 64 + h * 16 + d] = o[d];
}

// ---------------- output gating + mask blend ----------------
__global__ void k_final(const float* __restrict__ om, const float* __restrict__ gw,
                        const float* __restrict__ gb, const float* __restrict__ fw,
                        const float* __restrict__ fb, const float* __restrict__ hnew,
                        const float* __restrict__ hx, const float* __restrict__ cx,
                        const float* __restrict__ mask, float* __restrict__ out) {
  long idx = (long)blockIdx.x * blockDim.x + threadIdx.x;
  if (idx >= (long)B_ * NHID) return;
  int b = (int)(idx >> 11);
  int j = (int)(idx & 2047);
  int k = j >> 8, o = j & 255;
  const float* ov = om + (long)b * 512 + k * 64;
  float g = gb[o], f = fb[o];
#pragma unroll 4
  for (int d = 0; d < 64; ++d) {
    float x = ov[d];
    g += x * gw[d * 256 + o];
    f += x * fw[d * 256 + o];
  }
  g = 1.f / (1.f + __expf(-g));
  f = tanhf(f);
  float h = hnew[idx] + g * f;
  float m = mask[(long)b * 8 + k];
  out[idx]                    = m * h + (1.f - m) * hx[idx];
  out[(long)B_ * NHID + idx]  = m * h + (1.f - m) * cx[idx];
  out[2L * B_ * NHID + idx]   = m;
}

extern "C" void kernel_launch(void* const* d_in, const int* in_sizes, int n_in,
                              void* d_out, int out_size, void* d_ws, size_t ws_size,
                              hipStream_t stream) {
  (void)in_sizes; (void)n_in; (void)out_size; (void)ws_size;
  const float* inp     = (const float*)d_in[0];
  const float* hx      = (const float*)d_in[1];
  const float* cx      = (const float*)d_in[2];
  const float* Wq_i    = (const float*)d_in[4];
  const float* Wk_i    = (const float*)d_in[5];
  const float* Wv_i    = (const float*)d_in[6];
  const float* fc_i_w  = (const float*)d_in[7];
  const float* fc_i_b  = (const float*)d_in[8];
  const float* Wq_m    = (const float*)d_in[9];
  const float* Wk_m    = (const float*)d_in[10];
  const float* Wv_m    = (const float*)d_in[11];
  const float* fc_m_w  = (const float*)d_in[12];
  const float* fc_m_b  = (const float*)d_in[13];
  const float* gate_m_w= (const float*)d_in[14];
  const float* gate_m_b= (const float*)d_in[15];
  const float* Wih     = (const float*)d_in[16];
  const float* Whh     = (const float*)d_in[17];
  const float* bih     = (const float*)d_in[18];
  const float* bhh     = (const float*)d_in[19];
  float* out = (float*)d_out;

  char* ws = (char*)d_ws;
  size_t off = 0;
  auto alloc = [&](size_t bytes) { size_t o = off; off += (bytes + 255) & ~(size_t)255; return o; };
  u16* inp_b  = (u16*)(ws + alloc((size_t)B_ * NINP * 2));
  u16* hx_b   = (u16*)(ws + alloc((size_t)B_ * NHID * 2));
  u16* v1_b   = (u16*)(ws + alloc((size_t)B_ * 1024 * 2));
  u16* u_b    = (u16*)(ws + alloc((size_t)B_ * 1024 * 2));
  u16* hnew_b = (u16*)(ws + alloc((size_t)B_ * NHID * 2));
  u16* wv1t   = (u16*)(ws + alloc((size_t)1024 * 512 * 2));
  u16* fct    = (u16*)(ws + alloc((size_t)1024 * 1024 * 2));
  u16* wih_b  = (u16*)(ws + alloc((size_t)NBO * 768 * 1024 * 2));
  u16* whh_b  = (u16*)(ws + alloc((size_t)NBO * 768 * 256 * 2));
  u16* wm_b   = (u16*)(ws + alloc((size_t)NBO * 192 * 256 * 2));
  float* k1   = (float*)(ws + alloc((size_t)B_ * 64 * 4));
  float* iatt = (float*)(ws + alloc((size_t)B_ * 8 * 4));
  float* mask = (float*)(ws + alloc((size_t)B_ * 8 * 4));
  float* cko  = (float*)(ws + alloc((size_t)NBO * 768 * 4));
  float* hnew = (float*)(ws + alloc((size_t)B_ * NHID * 4));
  float* qkv  = (float*)(ws + alloc((size_t)B_ * NBO * 192 * 4));
  float* om   = (float*)(ws + alloc((size_t)B_ * 512 * 4));

  // conversions
  k_cvt<<<(B_ * NINP + 255) / 256, 256, 0, stream>>>(inp, inp_b, B_ * NINP);
  k_cvt<<<(B_ * NHID + 255) / 256, 256, 0, stream>>>(hx, hx_b, B_ * NHID);
  k_cvt<<<(NBO * 768 * 1024 + 255) / 256, 256, 0, stream>>>(Wih, wih_b, NBO * 768 * 1024);
  k_cvt<<<(NBO * 768 * 256 + 255) / 256, 256, 0, stream>>>(Whh, whh_b, NBO * 768 * 256);
  k_cvt_t<<<(512 * 1024 + 255) / 256, 256, 0, stream>>>(Wv_i + 512 * 1024, wv1t, 512, 1024);
  k_cvt_t<<<(1024 * 1024 + 255) / 256, 256, 0, stream>>>(fc_i_w, fct, 1024, 1024);
  k_build_wm<<<(NBO * 192 * 256 + 255) / 256, 256, 0, stream>>>(Wq_m, Wk_m, Wv_m, wm_b);
  k_cko<<<(NBO * 768 + 255) / 256, 256, 0, stream>>>(Wih, bih, fc_i_b, cko);

  // scores / mask
  k_k1<<<(B_ * 64) / 256, 256, 0, stream>>>(inp, Wk_i, k1);
  k_score<<<B_ / 32, 256, 0, stream>>>(hx, Wq_i, k1, iatt, mask);

  // v1 = inp @ Wv_i[1]   (M=4096, N=1024, K=512) -> bf16
  k_gemm<<<dim3(B_ / 64, 1024 / 256, 1), 256, 0, stream>>>(
      inp_b, NINP, 0, wv1t, 0, 512, nullptr, v1_b, 1024, 0, 512 / 32);
  // u = v1 @ fc_i_w      (M=4096, N=1024, K=1024) -> bf16
  k_gemm<<<dim3(B_ / 64, 1024 / 256, 1), 256, 0, stream>>>(
      v1_b, 1024, 0, fct, 0, 1024, nullptr, u_b, 1024, 0, 1024 / 32);

  // fused GRU -> h_new (f32 + bf16)
  k_gru<<<dim3(B_ / 16, NBO), 256, 0, stream>>>(
      u_b, hx_b, wih_b, whh_b, cko, bhh, iatt, hx, hnew, hnew_b);

  // QKV: per block k, (B x 256) @ (256 x 192) -> qkv f32
  // block of 192 threads = 6 waves, each wave covers 32 cols -> all 192 cols
  k_gemm<<<dim3(B_ / 64, 1, NBO), 192, 0, stream>>>(
      hnew_b, NHID, 256, wm_b, (long)192 * 256, 256, qkv, nullptr, NBO * 192, 192, 256 / 32);

  // mini attention -> om
  k_mattn<<<B_ / 8, 256, 0, stream>>>(qkv, om);

  // gating + mask blend -> outputs
  k_final<<<(int)(((long)B_ * NHID) / 256), 256, 0, stream>>>(
      om, gate_m_w, gate_m_b, fc_m_w, fc_m_b, hnew, hx, cx, mask, out);
}
